// ISTFT_29592324669660
// MI455X (gfx1250) — matrix operand order repaired and checked by reading, not compiled, not run on hardware
//
#include <hip/hip_runtime.h>
#include <math.h>

// ---------------- problem constants ----------------
#define N_FFT     1024
#define KSPEC     513          // rfft bins
#define KTILES    17           // K padded to 544 = 17*32
#define HOP       256
#define WIN       1024
#define PADL      384          // (WIN-HOP)/2
#define BATCH     16
#define T_FRAMES  2048
#define OUT_LEN   524288       // (2047*256+1024) - 2*384  (== 1<<19)
#define TOTAL_OUT (BATCH * OUT_LEN)

#define MTILES    64           // 1024/16 M-tiles
#define NBAS      (KTILES * MTILES * 32 * 16)   // bf16 elems per basis array

#define N_BLK     128          // frames per workgroup (8 N-tiles)
#define NT_PER_WG 8

typedef __attribute__((ext_vector_type(16))) __bf16 v16bf;
typedef __attribute__((ext_vector_type(8)))  float  v8f;

// ---------------- kernel 1: zero output accumulator ----------------
__global__ void istft_zero_kernel(float* __restrict__ out) {
  size_t i = (size_t)blockIdx.x * blockDim.x + threadIdx.x;
  if (i < (size_t)TOTAL_OUT) out[i] = 0.0f;
}

// ---------------- kernel 2: build windowed iDFT basis ----------------
// Bc[n,k] =  win[n] * (wk/N) * cos(2*pi*n*k/N)
// Bs[n,k] = -win[n] * (wk/N) * sin(2*pi*n*k/N),  wk = (k==0||k==512)?1:2
// Stored as bf16 hi/lo pairs directly in this code's contiguous-k 16x32 A-fragment lane
// layout: lane l -> M = l%16, kbase = (l/16)*16;
// element e (0..15) -> K = kt*32 + kbase + e.
__global__ void istft_basis_kernel(const float* __restrict__ win,
                                   __bf16* __restrict__ cosHi, __bf16* __restrict__ cosLo,
                                   __bf16* __restrict__ sinHi, __bf16* __restrict__ sinLo) {
  int gid = blockIdx.x * blockDim.x + threadIdx.x;
  if (gid >= 2 * KTILES * MTILES * 32) return;
  int lane = gid & 31;
  int r    = gid >> 5;
  int mt   = r & 63; r >>= 6;
  int kt   = r % KTILES;
  int mat  = r / KTILES;                 // 0 = cos, 1 = sin
  int n    = mt * 16 + (lane & 15);
  int kbase = (lane >> 4) * 8;
  float wn = win[n] * (1.0f / (float)N_FFT);

  v16bf hv, lv;
  #pragma unroll
  for (int e = 0; e < 16; ++e) {
    int k = kt * 32 + (lane >> 4) * 16 + e;
    float val = 0.0f;
    if (k < KSPEC) {
      float wk = (k == 0 || k == 512) ? 1.0f : 2.0f;
      int m = (n * k) & (N_FFT - 1);     // exact angle reduction
      float ang = (float)m * (6.283185307179586f / (float)N_FFT);
      float s, c;
      sincosf(ang, &s, &c);
      val = (mat == 0) ? (wn * wk * c) : (-wn * wk * s);
    }
    __bf16 h = (__bf16)val;
    __bf16 l = (__bf16)(val - (float)h); // residual -> second bf16 term
    hv[e] = h; lv[e] = l;
  }
  size_t off = ((size_t)(kt * MTILES + mt) * 32 + lane) * 16;
  if (mat == 0) { *(v16bf*)(cosHi + off) = hv; *(v16bf*)(cosLo + off) = lv; }
  else          { *(v16bf*)(sinHi + off) = hv; *(v16bf*)(sinLo + off) = lv; }
}

// ---------------- kernel 3: WMMA GEMM + windowed overlap-add ----------------
#define LDS_STRIDE 48   // bf16 elems per LDS row (32 used): rows stay 32B aligned

__global__ __launch_bounds__(256)
void istft_wmma_kernel(const float* __restrict__ spec_r, const float* __restrict__ spec_i,
                       const __bf16* __restrict__ cosHi, const __bf16* __restrict__ cosLo,
                       const __bf16* __restrict__ sinHi, const __bf16* __restrict__ sinLo,
                       float* __restrict__ out) {
  // staged spec tile, transposed [t][k], split bf16 hi/lo: 4 * 128*48*2B = 48 KB
  __shared__ __align__(32) __bf16 sReHi[N_BLK * LDS_STRIDE];
  __shared__ __align__(32) __bf16 sReLo[N_BLK * LDS_STRIDE];
  __shared__ __align__(32) __bf16 sImHi[N_BLK * LDS_STRIDE];
  __shared__ __align__(32) __bf16 sImLo[N_BLK * LDS_STRIDE];

  const int tid  = threadIdx.x;
  const int lane = tid & 31;
  const int w    = tid >> 5;                 // wave id 0..7
  // M-blocks on grid.x (fastest) so the 8 WGs sharing a spec tile run
  // back-to-back and hit L2; spec crosses HBM ~once.
  const int mt_g = blockIdx.x * 8 + w;       // this wave's M-tile (0..63)
  const int t0   = blockIdx.y * N_BLK;       // frame-column block
  const int b    = blockIdx.z;               // batch

  v8f acc[NT_PER_WG];
  #pragma unroll
  for (int nt = 0; nt < NT_PER_WG; ++nt) { v8f z = {}; acc[nt] = z; }

  // ---- software-pipelined staging registers (32(K) x 128(T) tile) ----
  // 1024 float4 per array; 256 threads -> 4 float4 each
  float4 vr[4], vi[4];
  auto load_tile = [&](int kt) {
    #pragma unroll
    for (int it = 0; it < 4; ++it) {
      int lin = tid + it * 256;              // 0..1023
      int kk  = lin >> 5;                    // 0..31 (k within tile)
      int t4  = lin & 31;                    // float4 column
      int kg  = kt * 32 + kk;
      float4 zr = make_float4(0.f, 0.f, 0.f, 0.f);
      vr[it] = zr; vi[it] = zr;
      if (kg < KSPEC) {
        size_t base = ((size_t)b * KSPEC + kg) * T_FRAMES + t0 + t4 * 4;
        vr[it] = *(const float4*)(spec_r + base);
        vi[it] = *(const float4*)(spec_i + base);
      }
    }
  };

  load_tile(0);   // prologue

  for (int kt = 0; kt < KTILES; ++kt) {
    __syncthreads();   // previous iteration done reading LDS
    // ---- convert staged registers -> bf16 hi/lo in LDS (transposed [t][k])
    #pragma unroll
    for (int it = 0; it < 4; ++it) {
      int lin = tid + it * 256;
      int kk  = lin >> 5;
      int t4  = lin & 31;
      float xr[4] = {vr[it].x, vr[it].y, vr[it].z, vr[it].w};
      float xi[4] = {vi[it].x, vi[it].y, vi[it].z, vi[it].w};
      #pragma unroll
      for (int j = 0; j < 4; ++j) {
        int idx = (t4 * 4 + j) * LDS_STRIDE + kk;
        __bf16 h = (__bf16)xr[j];
        sReHi[idx] = h;
        sReLo[idx] = (__bf16)(xr[j] - (float)h);
        h = (__bf16)xi[j];
        sImHi[idx] = h;
        sImLo[idx] = (__bf16)(xi[j] - (float)h);
      }
    }
    // ---- issue next tile's global loads now; latency overlaps the WMMAs below
    if (kt + 1 < KTILES) load_tile(kt + 1);
    __syncthreads();

    // ---- A fragments (pre-packed lane layout, one 32B load each, L2-resident)
    size_t aoff = ((size_t)(kt * MTILES + mt_g) * 32 + lane) * 16;
    v16bf aCh = *(const v16bf*)(cosHi + aoff);
    v16bf aCl = *(const v16bf*)(cosLo + aoff);
    v16bf aSh = *(const v16bf*)(sinHi + aoff);
    v16bf aSl = *(const v16bf*)(sinLo + aoff);

    // ---- 8 N-tiles x (split-bf16 3-product) x (cos*Re + sin*Im) = 48 WMMAs
    #pragma unroll
    for (int nt = 0; nt < NT_PER_WG; ++nt) {
      int trow = nt * 16 + (lane & 15);      // N = lane%16
      int kh   = (lane >> 4) * 16;           // lanes 0-15: K 0-15; 16-31: K 16-31
      v16bf bRh = *(const v16bf*)&sReHi[trow * LDS_STRIDE + kh];
      v16bf bRl = *(const v16bf*)&sReLo[trow * LDS_STRIDE + kh];
      v16bf bIh = *(const v16bf*)&sImHi[trow * LDS_STRIDE + kh];
      v16bf bIl = *(const v16bf*)&sImLo[trow * LDS_STRIDE + kh];
      acc[nt] = __builtin_amdgcn_wmma_f32_16x16x32_bf16(false, aCh, false, bRh, (short)0, acc[nt], false, false);
      acc[nt] = __builtin_amdgcn_wmma_f32_16x16x32_bf16(false, aCh, false, bRl, (short)0, acc[nt], false, false);
      acc[nt] = __builtin_amdgcn_wmma_f32_16x16x32_bf16(false, aCl, false, bRh, (short)0, acc[nt], false, false);
      acc[nt] = __builtin_amdgcn_wmma_f32_16x16x32_bf16(false, aSh, false, bIh, (short)0, acc[nt], false, false);
      acc[nt] = __builtin_amdgcn_wmma_f32_16x16x32_bf16(false, aSh, false, bIl, (short)0, acc[nt], false, false);
      acc[nt] = __builtin_amdgcn_wmma_f32_16x16x32_bf16(false, aSl, false, bIh, (short)0, acc[nt], false, false);
    }
  }

  // ---- epilogue: overlap-add windowed samples into trimmed output
  // C/D layout: VGPR v -> M = v + 8*(lane>=16); N = lane%16
  const int  mrow  = mt_g * 16 + ((lane >> 4) << 3);
  const int  tcol0 = t0 + (lane & 15);
  float* outb = out + (size_t)b * OUT_LEN;
  #pragma unroll
  for (int nt = 0; nt < NT_PER_WG; ++nt) {
    long p0 = (long)(tcol0 + nt * 16) * HOP + mrow - PADL;
    #pragma unroll
    for (int v = 0; v < 8; ++v) {
      long p = p0 + v;
      if (p >= 0 && p < (long)OUT_LEN) atomicAdd(outb + p, acc[nt][v]);
    }
  }
}

// ---------------- kernel 4: divide by analytic window envelope ----------------
__global__ void istft_norm_kernel(float* __restrict__ out, const float* __restrict__ win) {
  size_t gid = (size_t)blockIdx.x * blockDim.x + threadIdx.x;
  if (gid >= (size_t)TOTAL_OUT) return;
  int pp = (int)(gid & (OUT_LEN - 1));      // position within batch (OUT_LEN is 2^19)
  int pu = pp + PADL;                        // untrimmed position
  int jhi = pu >> 8; if (jhi > T_FRAMES - 1) jhi = T_FRAMES - 1;
  int jlo = pu - (WIN - HOP);                // ceil((pu-1023)/256) == floor((pu-768)/256)
  jlo = (jlo < 0) ? 0 : (jlo >> 8);
  float env = 0.0f;
  for (int j = jlo; j <= jhi; ++j) {         // at most 4 overlapping frames
    float wv = win[pu - j * HOP];
    env += wv * wv;
  }
  env = fmaxf(env, 1e-11f);
  out[gid] = out[gid] / env;
}

// ---------------- launcher ----------------
extern "C" void kernel_launch(void* const* d_in, const int* in_sizes, int n_in,
                              void* d_out, int out_size, void* d_ws, size_t ws_size,
                              hipStream_t stream) {
  const float* spec_r = (const float*)d_in[0];
  const float* spec_i = (const float*)d_in[1];
  const float* window = (const float*)d_in[2];
  float* out = (float*)d_out;

  __bf16* ws    = (__bf16*)d_ws;            // 4 * NBAS bf16 ~ 4.5 MB scratch
  __bf16* cosHi = ws;
  __bf16* cosLo = ws + (size_t)NBAS;
  __bf16* sinHi = ws + (size_t)2 * NBAS;
  __bf16* sinLo = ws + (size_t)3 * NBAS;

  // 1) zero accumulation buffer (harness poisons d_out)
  istft_zero_kernel<<<(TOTAL_OUT + 255) / 256, 256, 0, stream>>>(out);

  // 2) build windowed iDFT basis fragments (cheap: ~1.1M sincos)
  int basisThreads = 2 * KTILES * MTILES * 32;
  istft_basis_kernel<<<(basisThreads + 255) / 256, 256, 0, stream>>>(
      window, cosHi, cosLo, sinHi, sinLo);

  // 3) iDFT-as-GEMM via v_wmma_f32_16x16x32_bf16 + atomic overlap-add
  //    grid.x = M-blocks (fastest) -> spec tiles reused out of L2
  dim3 grid(N_FFT / 128, T_FRAMES / N_BLK, BATCH);   // (8, 16, 16)
  istft_wmma_kernel<<<grid, 256, 0, stream>>>(
      spec_r, spec_i, cosHi, cosLo, sinHi, sinLo, out);

  // 4) divide by window^2 envelope (analytic, no atomics)
  istft_norm_kernel<<<(TOTAL_OUT + 255) / 256, 256, 0, stream>>>(out, window);
}
